// InteractionNetwork_79199196938367
// MI455X (gfx1250) — compile-verified, matrix operand order checked by reading, not run on hardware
//
#include <hip/hip_runtime.h>
#include <hip/hip_bf16.h>

#define N_NODES   100000
#define N_EDGESN  1200000
#define E_TOT     (N_EDGESN + N_NODES)   // self loops appended
#define HID       64
#define HEADS     4
#define CH        16
#define N_GRAPHS  64
#define NEG_SLOPE 0.2f

typedef __attribute__((ext_vector_type(16))) _Float16 v16h;
typedef __attribute__((ext_vector_type(8)))  float    v8f;

// ---- order-preserving float<->uint encoding for atomic max over floats ----
__device__ __forceinline__ unsigned encf(float f) {
  unsigned u = __float_as_uint(f);
  return (u & 0x80000000u) ? ~u : (u | 0x80000000u);
}
__device__ __forceinline__ float decf(unsigned u) {
  return __uint_as_float((u & 0x80000000u) ? (u ^ 0x80000000u) : ~u);
}

// ---------------------------------------------------------------------------
// Y[nrows,64] = X[nrows,64] @ W[64,64] + bias, via v_wmma_f32_16x16x32_f16.
// One wave per 16x16 output tile; K=64 done as two K=32 WMMA steps.
// Block = 128 threads (4 waves) -> 4 column tiles = all 64 output columns.
// grid.x = nrows/16 (100000/16 = 6250 exactly; EXEC stays all-ones).
// ---------------------------------------------------------------------------
__global__ void gemm64_wmma(const float* __restrict__ X,
                            const float* __restrict__ W,
                            const float* __restrict__ bias,
                            float* __restrict__ Y) {
  const int wave  = threadIdx.x >> 5;
  const int lane  = threadIdx.x & 31;
  const int m     = lane & 15;   // row (A) / column (B,C)
  const int hi    = lane >> 4;   // K-half / M-half selector
  const int rbase = blockIdx.x * 16;
  const int cbase = wave * 16;

  v16h a0, a1, b0, b1;
  const float* xrow = X + (size_t)(rbase + m) * HID;
  // 16-bit A 16x32 layout: lanes 0-15 hold K {0..7,16..23}, lanes 16-31 hold
  // K {8..15,24..31}; half j -> K = (j>>3)*16 + hi*8 + (j&7). B mirrors with
  // N=lane column. a1/b1 are the K=32..63 half.
  #pragma unroll
  for (int j = 0; j < 16; ++j) {
    const int k = ((j >> 3) << 4) + (hi << 3) + (j & 7);
    a0[j] = (_Float16)xrow[k];
    a1[j] = (_Float16)xrow[32 + k];
    b0[j] = (_Float16)W[(size_t)k * HID + cbase + m];
    b1[j] = (_Float16)W[(size_t)(32 + k) * HID + cbase + m];
  }

  const float bv = bias[cbase + m];
  v8f acc;
  #pragma unroll
  for (int r = 0; r < 8; ++r) acc[r] = bv;   // bias per output column

  acc = __builtin_amdgcn_wmma_f32_16x16x32_f16(false, a0, false, b0,
                                               (short)0, acc, false, false);
  acc = __builtin_amdgcn_wmma_f32_16x16x32_f16(false, a1, false, b1,
                                               (short)0, acc, false, false);

  // C/D layout: lane -> column N, VGPR r -> row M (hi selects M+8).
  #pragma unroll
  for (int r = 0; r < 8; ++r)
    Y[(size_t)(rbase + (hi << 3) + r) * HID + cbase + m] = acc[r];
}

// ---------------------------------------------------------------------------
__global__ void zero_graph(float* __restrict__ graph) {
  int t = blockIdx.x * blockDim.x + threadIdx.x;
  if (t < N_GRAPHS * HID) graph[t] = 0.f;
}

__global__ void init_layer(unsigned* __restrict__ smax,
                           float* __restrict__ den,
                           float* __restrict__ agg) {
  int t = blockIdx.x * blockDim.x + threadIdx.x;
  if (t < N_NODES * HEADS) { smax[t] = 0u; den[t] = 0.f; }
  if (t < N_NODES * HID)   agg[t] = 0.f;
}

__device__ __forceinline__ void edge_ids(int e, const int* __restrict__ ei,
                                         int& s, int& d) {
  if (e < N_EDGESN) { s = ei[e]; d = ei[N_EDGESN + e]; }
  else              { s = d = e - N_EDGESN; }          // self loop
}

// Pass A: score[e,h] = sum_c att[h,c]*leakyrelu(xl[s]+xr[d]); atomic max per dst.
__global__ void edge_score_max(const float* __restrict__ xl,
                               const float* __restrict__ xr,
                               const float* __restrict__ att,
                               const int* __restrict__ ei,
                               float* __restrict__ score,
                               unsigned* __restrict__ smax) {
  int t = blockIdx.x * blockDim.x + threadIdx.x;
  if (t >= E_TOT * HEADS) return;
  const int e = t >> 2, h = t & 3;
  int s, d; edge_ids(e, ei, s, d);
  const float* pl = xl + (size_t)s * HID + h * CH;
  const float* pr = xr + (size_t)d * HID + h * CH;
  const float* pa = att + h * CH;
  float sc = 0.f;
  #pragma unroll
  for (int c = 0; c < CH; ++c) {
    float v = pl[c] + pr[c];
    v = v > 0.f ? v : v * NEG_SLOPE;
    sc += v * pa[c];
  }
  score[t] = sc;
  atomicMax(&smax[d * HEADS + h], encf(sc));
}

// Pass B: ex = exp(score - max); accumulate denominator per dst.
__global__ void edge_exp_sum(const int* __restrict__ ei,
                             float* __restrict__ score,
                             const unsigned* __restrict__ smax,
                             float* __restrict__ den) {
  int t = blockIdx.x * blockDim.x + threadIdx.x;
  if (t >= E_TOT * HEADS) return;
  const int e = t >> 2, h = t & 3;
  int s, d; edge_ids(e, ei, s, d);
  const float ex = __expf(score[t] - decf(smax[d * HEADS + h]));
  score[t] = ex;
  atomicAdd(&den[d * HEADS + h], ex);
}

// Pass C: agg[dst] += (ex/den) * xl[src]  (16 lanes of the head).
__global__ void edge_aggregate(const int* __restrict__ ei,
                               const float* __restrict__ score,
                               const float* __restrict__ den,
                               const float* __restrict__ xl,
                               float* __restrict__ agg) {
  int t = blockIdx.x * blockDim.x + threadIdx.x;
  if (t >= E_TOT * HEADS) return;
  const int e = t >> 2, h = t & 3;
  int s, d; edge_ids(e, ei, s, d);
  const float alpha = score[t] / den[d * HEADS + h];
  const float* pl = xl + (size_t)s * HID + h * CH;
  float* pd = agg + (size_t)d * HID + h * CH;
  #pragma unroll
  for (int c = 0; c < CH; ++c) atomicAdd(&pd[c], alpha * pl[c]);
}

// Finalize: x = relu(agg + bias); layer 2 also segment-sums into graph output.
__global__ void node_finalize(const float* __restrict__ agg,
                              const float* __restrict__ bias,
                              float* __restrict__ xout,
                              const int* __restrict__ batch,
                              float* __restrict__ graph,
                              int do_graph) {
  int t = blockIdx.x * blockDim.x + threadIdx.x;
  if (t >= N_NODES * HID) return;
  const int node = t >> 6, col = t & 63;
  float v = agg[t] + bias[col];
  v = v > 0.f ? v : 0.f;
  xout[t] = v;
  if (do_graph) atomicAdd(&graph[(size_t)batch[node] * HID + col], v);
}

// ---------------------------------------------------------------------------
extern "C" void kernel_launch(void* const* d_in, const int* in_sizes, int n_in,
                              void* d_out, int out_size, void* d_ws, size_t ws_size,
                              hipStream_t stream) {
  const float* x0    = (const float*)d_in[0];
  const int*   ei    = (const int*)d_in[1];
  const int*   batch = (const int*)d_in[2];

  float* out   = (float*)d_out;
  float* xout  = out;                            // [N_NODES*HID] final/intermediate x
  float* graph = out + (size_t)N_NODES * HID;    // [64*64]

  // workspace carve-up (~75 MB)
  char* p = (char*)d_ws;
  float*    xl    = (float*)p;    p += sizeof(float) * (size_t)N_NODES * HID;
  float*    xr    = (float*)p;    p += sizeof(float) * (size_t)N_NODES * HID;
  float*    score = (float*)p;    p += sizeof(float) * (size_t)E_TOT * HEADS;
  float*    den   = (float*)p;    p += sizeof(float) * (size_t)N_NODES * HEADS;
  unsigned* smax  = (unsigned*)p;

  const int EB = (E_TOT * HEADS + 255) / 256;        // edge-head grid
  const int NB = (N_NODES * HID + 255) / 256;        // node-col grid
  const int GEMM_B = N_NODES / 16;                   // 6250

  zero_graph<<<(N_GRAPHS * HID + 255) / 256, 256, 0, stream>>>(graph);

  for (int L = 0; L < 2; ++L) {
    const float* xin = (L == 0) ? x0 : xout;
    const float* Wl  = (const float*)d_in[3 + L * 6 + 0];
    const float* bl  = (const float*)d_in[3 + L * 6 + 1];
    const float* Wr  = (const float*)d_in[3 + L * 6 + 2];
    const float* br  = (const float*)d_in[3 + L * 6 + 3];
    const float* att = (const float*)d_in[3 + L * 6 + 4];
    const float* b   = (const float*)d_in[3 + L * 6 + 5];

    gemm64_wmma<<<GEMM_B, 128, 0, stream>>>(xin, Wl, bl, xl);
    gemm64_wmma<<<GEMM_B, 128, 0, stream>>>(xin, Wr, br, xr);

    // agg aliases xout: GEMMs above already consumed previous x.
    init_layer<<<NB, 256, 0, stream>>>(smax, den, xout);

    edge_score_max<<<EB, 256, 0, stream>>>(xl, xr, att, ei, score, smax);
    edge_exp_sum  <<<EB, 256, 0, stream>>>(ei, score, smax, den);
    edge_aggregate<<<EB, 256, 0, stream>>>(ei, score, den, xl, xout);

    node_finalize<<<NB, 256, 0, stream>>>(xout, b, xout, batch, graph, L == 1);
  }
}
